// SelectiveRecurrentAttention_11725260718450
// MI455X (gfx1250) — compile-verified, matrix operand order checked by reading, not run on hardware
//
#include <hip/hip_runtime.h>
#include <hip/hip_bf16.h>

// ---------------------------------------------------------------------------
// SelectiveRecurrentAttention for gfx1250 (MI455X), bf16 WMMA pipeline.
// Async-to-LDS staging (ASYNCcnt) + WMMA everywhere that matters.
// ---------------------------------------------------------------------------

typedef __attribute__((ext_vector_type(16))) __bf16 v16bf;
typedef __attribute__((ext_vector_type(8)))  float  v8f;
typedef __attribute__((ext_vector_type(4)))  int    v4i;

#define BATCH    4
#define SEQ_N    1024
#define DMODEL   1024
#define H_HEADS  16
#define HD       64
#define QKV_N    3072
#define SCALE_QK 0.125f   // 64^-0.5

#if __has_builtin(__builtin_amdgcn_global_load_async_to_lds_b128)
#define HAVE_ASYNC_LDS 1
#endif

// 16-byte global -> LDS copy. Async (no VGPR round-trip) when available.
// Builtin signature: (v4i AS1* gsrc, v4i AS3* ldst, imm offset, imm cpol).
__device__ __forceinline__ void cp16_g2l(void* l, const void* g) {
#ifdef HAVE_ASYNC_LDS
  __builtin_amdgcn_global_load_async_to_lds_b128(
      (__attribute__((address_space(1))) v4i*)(uintptr_t)g,
      (__attribute__((address_space(3))) v4i*)(unsigned int)(uintptr_t)l,
      0, 0);
#else
  *(uint4*)l = *(const uint4*)g;
#endif
}

__device__ __forceinline__ void wait_async_copies() {
#ifdef HAVE_ASYNC_LDS
  asm volatile("s_wait_asynccnt 0" ::: "memory");
#endif
}

// ---------------------------------------------------------------------------
// Fragment loader: 16x32 bf16 operand (A layout; B mirrored N<->M roles).
//   v0..3 : K = v*2 + half*8 ; v4..7 : K = 16 + (v-4)*2 + half*8
// Pairs contiguous in K -> aligned 32-bit LDS reads (compiler merges to b128).
// ---------------------------------------------------------------------------
__device__ __forceinline__ v16bf load_frag16x32(const __bf16* base, int stride,
                                                int r, int half) {
  v16bf f;
  const __bf16* row = base + r * stride;
#pragma unroll
  for (int v = 0; v < 8; ++v) {
    const int kb = (v < 4) ? (v * 2 + half * 8) : (16 + (v - 4) * 2 + half * 8);
    f[2 * v]     = row[kb];
    f[2 * v + 1] = row[kb + 1];
  }
  return f;
}

// ---------------------------------------------------------------------------
// f32 -> bf16 convert (grid-stride)
// ---------------------------------------------------------------------------
__global__ void cvt_kernel(const float* __restrict__ in, __bf16* __restrict__ out,
                           long n) {
  long i = (long)blockIdx.x * blockDim.x + threadIdx.x;
  long stride = (long)gridDim.x * blockDim.x;
  for (; i < n; i += stride) out[i] = (__bf16)in[i];
}

// f32[K][N] -> bf16[N][K] transpose-convert (weights, done once; K=1024).
__global__ void cvt_tr_kernel(const float* __restrict__ in, __bf16* __restrict__ out,
                              int K, int N) {
  long idx = (long)blockIdx.x * blockDim.x + threadIdx.x;
  const long total = (long)K * N;
  if (idx >= total) return;
  const int k = (int)(idx & (K - 1));   // K is a power of two (1024)
  const int n = (int)(idx >> 10);
  out[(size_t)n * K + k] = (__bf16)in[(size_t)k * N + n];
}

// ---------------------------------------------------------------------------
// bf16 GEMM: C[M,N] = A[M,K] * BT[N,K]^T, f32 accumulate.
// Block tile 128x128x32; 8 waves as 4(M) x 2(N); wave tile 32x64 -> 8 wmma.
// Both operands stored K-contiguous -> all staging is async b128 copies.
// ---------------------------------------------------------------------------
__global__ __launch_bounds__(256)
void gemm_bf16_kernel(const __bf16* __restrict__ A, const __bf16* __restrict__ BT,
                      float* __restrict__ C, int M, int N, int K) {
  __shared__ __align__(16) __bf16 As[128 * 40];   // [row][k], stride 40 (80B)
  __shared__ __align__(16) __bf16 Bs[128 * 40];   // [n][k]

  const int t    = threadIdx.x;
  const int lane = t & 31;
  const int wave = t >> 5;
  const int wy   = wave >> 1;      // 0..3 -> M offset wy*32
  const int wx   = wave & 1;       // 0..1 -> N offset wx*64
  const int cn   = lane & 15;
  const int ch   = lane >> 4;
  const int bm   = blockIdx.y * 128;
  const int bn   = blockIdx.x * 128;

  const int r0 = t >> 2;            // 0..63
  const int c0 = (t & 3) * 8;       // 0,8,16,24

  v8f acc[2][4] = {};

  for (int k0 = 0; k0 < K; k0 += 32) {
    // A tile 128x32: 512 b128 copies, 2 per thread
    cp16_g2l(As + r0 * 40 + c0,        A + (size_t)(bm + r0) * K + k0 + c0);
    cp16_g2l(As + (r0 + 64) * 40 + c0, A + (size_t)(bm + r0 + 64) * K + k0 + c0);
    // B tile 128x32 (from BT[N][K]): 2 per thread
    cp16_g2l(Bs + r0 * 40 + c0,        BT + (size_t)(bn + r0) * K + k0 + c0);
    cp16_g2l(Bs + (r0 + 64) * 40 + c0, BT + (size_t)(bn + r0 + 64) * K + k0 + c0);

    if (k0 + 32 < K) {  // global_prefetch_b8 next tiles
      __builtin_prefetch(A + (size_t)(bm + (t >> 1)) * K + k0 + 32 + (t & 1) * 16, 0, 1);
      __builtin_prefetch(BT + (size_t)(bn + (t >> 1)) * K + k0 + 32 + (t & 1) * 16, 0, 1);
    }
    wait_async_copies();
    __syncthreads();

    v16bf af[2], bfv[4];
#pragma unroll
    for (int i = 0; i < 2; ++i)
      af[i] = load_frag16x32(As + (wy * 32 + i * 16) * 40, 40, cn, ch);
#pragma unroll
    for (int j = 0; j < 4; ++j)
      bfv[j] = load_frag16x32(Bs + (wx * 64 + j * 16) * 40, 40, cn, ch);

#pragma unroll
    for (int i = 0; i < 2; ++i)
#pragma unroll
      for (int j = 0; j < 4; ++j)
        acc[i][j] = __builtin_amdgcn_wmma_f32_16x16x32_bf16(
            false, af[i], false, bfv[j], (short)0, acc[i][j], false, false);
    __syncthreads();
  }

  // C layout: lane 0-15 col n, vgpr g row (g + 8*half)
#pragma unroll
  for (int i = 0; i < 2; ++i)
#pragma unroll
    for (int j = 0; j < 4; ++j)
#pragma unroll
      for (int g = 0; g < 8; ++g) {
        const int row = bm + wy * 32 + i * 16 + g + ch * 8;
        const int col = bn + wx * 64 + j * 16 + cn;
        C[(size_t)row * N + col] = acc[i][j][g];
      }
}

// ---------------------------------------------------------------------------
// gamma = sigmoid(X @ W_gamma)^(1/20); log_b = cumsum(log(gamma+1e-8)) over n.
// One 1024-thread block per (b,h); Hillis-Steele inclusive scan in LDS.
// ---------------------------------------------------------------------------
__global__ __launch_bounds__(1024)
void gamma_scan_kernel(const float* __restrict__ X, const float* __restrict__ Wg,
                       float* __restrict__ LB) {
  const int bh = blockIdx.x;
  const int b  = bh >> 4;
  const int h  = bh & 15;
  const int n  = threadIdx.x;

  const float* xr = X + ((size_t)b * SEQ_N + n) * DMODEL;
  float z = 0.f;
  for (int d = 0; d < DMODEL; ++d) z += xr[d] * Wg[d * H_HEADS + h];

  const float sig = 1.f / (1.f + __expf(-z));
  const float gam = __powf(sig, 0.05f);                 // ^(1/TAU), TAU=20
  const float lg  = __logf(gam + 1e-8f);

  __shared__ float s[SEQ_N];
  s[n] = lg;
  __syncthreads();
  for (int off = 1; off < SEQ_N; off <<= 1) {
    const float p = (n >= off) ? s[n - off] : 0.f;
    __syncthreads();
    s[n] += p;
    __syncthreads();
  }
  LB[(size_t)bh * SEQ_N + n] = s[n];
}

// ---------------------------------------------------------------------------
// xpos rotate q,k -> (b,h,n,d) bf16; v -> TRANSPOSED (b,h,d,n) bf16 so the
// attention V-chunk staging is contiguous b128 async copies.
// ---------------------------------------------------------------------------
__global__ void rotate_kernel(const float* __restrict__ qkv,
                              const float* __restrict__ ts,
                              __bf16* __restrict__ Q, __bf16* __restrict__ Kt,
                              __bf16* __restrict__ Vt) {
  long idx = (long)blockIdx.x * blockDim.x + threadIdx.x;
  const long total = (long)BATCH * SEQ_N * H_HEADS * (HD / 2);
  if (idx >= total) return;

  const int dp = idx & 31;
  const int h  = (idx >> 5) & 15;
  const int n  = (idx >> 9) & 1023;
  const int b  = (int)(idx >> 19);
  const int d  = dp * 2;
  const int bh = b * H_HEADS + h;

  const float t    = ts[b * SEQ_N + n];
  const float invf = __powf(10000.f, -((float)d) / (float)HD);
  const float ang  = t * invf;
  const float sn = __sinf(ang), cs = __cosf(ang);
  const float sv = ((float)d + 0.4f * HD) / (1.4f * HD);
  const float scale = __powf(sv, t / 512.f);

  const float* base = qkv + ((size_t)b * SEQ_N + n) * QKV_N;
  const float q0 = base[h * HD + d],          q1 = base[h * HD + d + 1];
  const float k0 = base[1024 + h * HD + d],   k1 = base[1024 + h * HD + d + 1];
  const float v0 = base[2048 + h * HD + d],   v1 = base[2048 + h * HD + d + 1];

  const size_t o = (((size_t)bh) * SEQ_N + n) * HD + d;
  Q[o]      = (__bf16)((q0 * cs - q1 * sn) * scale);
  Q[o + 1]  = (__bf16)((q1 * cs + q0 * sn) * scale);
  Kt[o]     = (__bf16)((k0 * cs - k1 * sn) / scale);
  Kt[o + 1] = (__bf16)((k1 * cs + k0 * sn) / scale);

  const size_t ov = ((size_t)bh * HD + d) * SEQ_N + n;   // (b,h,d,n)
  Vt[ov]          = (__bf16)v0;
  Vt[ov + SEQ_N]  = (__bf16)v1;
}

// ---------------------------------------------------------------------------
// Tiled causal decay attention (per head): O = (QK^T * SCALE * D) V.
// Block = 128 q rows of one (b,h); 8 waves own 16-row q tiles.
// Key chunks of 32: 4 wmma for S, f32 decay/mask, per-wave LDS relayout
// (S -> A fragment, guarded by s_wait_dscnt), 4 wmma for O.
// ---------------------------------------------------------------------------
__global__ __launch_bounds__(256)
void attn_kernel(const __bf16* __restrict__ Q, const __bf16* __restrict__ Km,
                 const __bf16* __restrict__ Vm, const float* __restrict__ LB,
                 float* __restrict__ O) {
  __shared__ __align__(16) __bf16 Qs[128 * 72];      // [qrow][d]
  __shared__ __align__(16) __bf16 Kc[32 * 72];       // [key][d]  (B-frag for S)
  __shared__ __align__(16) __bf16 Vc[64 * 40];       // [d][key]  (B-frag for O)
  __shared__ __align__(16) __bf16 Sw[8 * 16 * 40];   // per-wave S tile
  __shared__ float Lq[128];
  __shared__ float Lk[32];

  const int t    = threadIdx.x;
  const int lane = t & 31;
  const int wave = t >> 5;
  const int cn   = lane & 15;
  const int ch   = lane >> 4;

  const int qbase = blockIdx.x * 128;
  const int bh    = blockIdx.z * H_HEADS + blockIdx.y;
  const size_t qoff = (size_t)bh * SEQ_N * HD;    // (b,h,n,d) base
  const size_t voff = (size_t)bh * HD * SEQ_N;    // (b,h,d,n) base

  // stage Q tile (128x64): 1024 b128 copies, 4 per thread
  {
    const int row = t >> 3;          // 0..31
    const int dc  = (t & 7) * 8;
#pragma unroll
    for (int j = 0; j < 4; ++j)
      cp16_g2l(Qs + (row + j * 32) * 72 + dc,
               Q + qoff + (size_t)(qbase + row + j * 32) * HD + dc);
  }
  if (t < 128) Lq[t] = LB[(size_t)bh * SEQ_N + qbase + t];
  wait_async_copies();
  __syncthreads();

  v16bf aq[2];
  aq[0] = load_frag16x32(Qs + (wave * 16) * 72 + 0,  72, cn, ch);
  aq[1] = load_frag16x32(Qs + (wave * 16) * 72 + 32, 72, cn, ch);

  v8f o[4] = {};
  const int q0 = qbase + wave * 16;

  for (int kc = 0; kc < qbase + 128; kc += 32) {
    // K chunk [32 keys][64 d]: 256 copies, 1/thread
    {
      const int key = t >> 3;
      const int dc  = (t & 7) * 8;
      cp16_g2l(Kc + key * 72 + dc,
               Km + qoff + (size_t)(kc + key) * HD + dc);
    }
    // V chunk [64 d][32 keys] from (b,h,d,n): 256 copies, 1/thread
    {
      const int dr = t >> 2;          // 0..63
      const int kk = (t & 3) * 8;     // 0,8,16,24
      cp16_g2l(Vc + dr * 40 + kk,
               Vm + voff + (size_t)dr * SEQ_N + kc + kk);
    }
    if (t < 32) Lk[t] = LB[(size_t)bh * SEQ_N + kc + t];
    wait_async_copies();
    __syncthreads();

    // S(16x32) = q . k^T  (accumulate over d=64 in two K=32 steps)
    v8f s[2] = {};
#pragma unroll
    for (int scol = 0; scol < 2; ++scol)
#pragma unroll
      for (int dc = 0; dc < 2; ++dc)
        s[scol] = __builtin_amdgcn_wmma_f32_16x16x32_bf16(
            false, aq[dc], false,
            load_frag16x32(Kc + (scol * 16) * 72 + dc * 32, 72, cn, ch),
            (short)0, s[scol], false, false);

    // decay + causal mask + scale, write bf16 S tile into per-wave LDS
    __bf16* swp = Sw + wave * (16 * 40);
#pragma unroll
    for (int scol = 0; scol < 2; ++scol)
#pragma unroll
      for (int g = 0; g < 8; ++g) {
        const int key = kc + scol * 16 + cn;
        const int qn  = q0 + g + ch * 8;
        const float dlt = Lq[qn - qbase] - Lk[scol * 16 + cn];
        const float w =
            (qn >= key) ? __expf(fminf(fmaxf(dlt, -80.f), 0.f)) : 0.f;
        swp[(g + ch * 8) * 40 + scol * 16 + cn] =
            (__bf16)(s[scol][g] * (SCALE_QK * w));
      }
    asm volatile("s_wait_dscnt 0" ::: "memory");  // per-wave LDS RAW

    // O(16x64) += S(16x32) @ V(32x64)
    const v16bf aS = load_frag16x32(swp, 40, cn, ch);
#pragma unroll
    for (int dt = 0; dt < 4; ++dt)
      o[dt] = __builtin_amdgcn_wmma_f32_16x16x32_bf16(
          false, aS, false, load_frag16x32(Vc + (dt * 16) * 40, 40, cn, ch),
          (short)0, o[dt], false, false);
    __syncthreads();
  }

#pragma unroll
  for (int dt = 0; dt < 4; ++dt)
#pragma unroll
    for (int g = 0; g < 8; ++g)
      O[qoff + (size_t)(q0 + g + ch * 8) * HD + dt * 16 + cn] = o[dt][g];
}

// ---------------------------------------------------------------------------
// Per-(b,h) mean / rstd over N*HD elements.
// ---------------------------------------------------------------------------
__global__ __launch_bounds__(256)
void stats_kernel(const float* __restrict__ O, float* __restrict__ st) {
  const int bh = blockIdx.x;
  const float* p = O + (size_t)bh * SEQ_N * HD;
  float s = 0.f, sq = 0.f;
  for (int i = threadIdx.x; i < SEQ_N * HD; i += 256) {
    const float v = p[i];
    s += v; sq += v * v;
  }
  __shared__ float ss[256], qs[256];
  ss[threadIdx.x] = s; qs[threadIdx.x] = sq;
  __syncthreads();
  for (int off = 128; off > 0; off >>= 1) {
    if (threadIdx.x < off) {
      ss[threadIdx.x] += ss[threadIdx.x + off];
      qs[threadIdx.x] += qs[threadIdx.x + off];
    }
    __syncthreads();
  }
  if (threadIdx.x == 0) {
    const float inv = 1.f / (float)(SEQ_N * HD);
    const float mu  = ss[0] * inv;
    const float var = qs[0] * inv - mu * mu;
    st[bh * 2]     = mu;
    st[bh * 2 + 1] = rsqrtf(var + 1e-5f);
  }
}

// ---------------------------------------------------------------------------
// groupnorm * silu(gated) -> bf16 in (b,n, h*64+d) layout for the out GEMM.
// ---------------------------------------------------------------------------
__global__ void norm_kernel(const float* __restrict__ Oat,
                            const float* __restrict__ gated,
                            const float* __restrict__ st,
                            __bf16* __restrict__ On) {
  long idx = (long)blockIdx.x * blockDim.x + threadIdx.x;
  const long total = (long)BATCH * SEQ_N * DMODEL;
  if (idx >= total) return;
  const int d = idx & 63;
  const int h = (idx >> 6) & 15;
  const int n = (idx >> 10) & 1023;
  const int b = (int)(idx >> 20);
  const int bh = b * H_HEADS + h;

  const float v  = Oat[(((size_t)bh) * SEQ_N + n) * HD + d];
  const float mu = st[bh * 2];
  const float rs = st[bh * 2 + 1];
  const float g  = gated[((size_t)b * SEQ_N + n) * DMODEL + h * HD + d];
  const float sil = g / (1.f + __expf(-g));
  On[((size_t)b * SEQ_N + n) * DMODEL + h * HD + d] =
      (__bf16)((v - mu) * rs * sil);
}

// ---------------------------------------------------------------------------
// Host launcher
// ---------------------------------------------------------------------------
extern "C" void kernel_launch(void* const* d_in, const int* in_sizes, int n_in,
                              void* d_out, int out_size, void* d_ws, size_t ws_size,
                              hipStream_t stream) {
  const float* X    = (const float*)d_in[0];
  const float* TS   = (const float*)d_in[1];
  const float* Wqkv = (const float*)d_in[2];
  const float* Wg   = (const float*)d_in[3];
  const float* Wgt  = (const float*)d_in[4];
  const float* Wo   = (const float*)d_in[5];
  float* out = (float*)d_out;

  char* ws = (char*)d_ws;
  size_t off = 0;
  auto take = [&](size_t bytes) -> void* {
    void* p = ws + off;
    off += (bytes + 255) & ~(size_t)255;
    return p;
  };
  const size_t MTOK = (size_t)BATCH * SEQ_N;  // 4096

  __bf16* Xb   = (__bf16*)take(MTOK * DMODEL * 2);
  __bf16* Wqb  = (__bf16*)take((size_t)DMODEL * QKV_N * 2);   // transposed [N][K]
  __bf16* Wgb  = (__bf16*)take((size_t)DMODEL * DMODEL * 2);  // transposed
  __bf16* Wob  = (__bf16*)take((size_t)DMODEL * DMODEL * 2);  // transposed
  float*  Cqkv = (float*)take(MTOK * QKV_N * 4);
  float*  Cgat = (float*)take(MTOK * DMODEL * 4);
  __bf16* Qr   = (__bf16*)take(MTOK * DMODEL * 2);
  __bf16* Kr   = (__bf16*)take(MTOK * DMODEL * 2);
  __bf16* Vr   = (__bf16*)take(MTOK * DMODEL * 2);            // (b,h,d,n)
  float*  LB   = (float*)take((size_t)BATCH * H_HEADS * SEQ_N * 4);
  float*  Oat  = (float*)take(MTOK * DMODEL * 4);
  float*  ST   = (float*)take((size_t)BATCH * H_HEADS * 2 * 4);
  __bf16* On   = (__bf16*)take(MTOK * DMODEL * 2);

  // 1) downconvert X; transpose-downconvert weights (K=1024 rows each)
  cvt_kernel<<<1024, 256, 0, stream>>>(X, Xb, (long)MTOK * DMODEL);
  cvt_tr_kernel<<<(int)(((size_t)DMODEL * QKV_N + 255) / 256), 256, 0, stream>>>(
      Wqkv, Wqb, DMODEL, QKV_N);
  cvt_tr_kernel<<<(int)(((size_t)DMODEL * DMODEL + 255) / 256), 256, 0, stream>>>(
      Wgt, Wgb, DMODEL, DMODEL);
  cvt_tr_kernel<<<(int)(((size_t)DMODEL * DMODEL + 255) / 256), 256, 0, stream>>>(
      Wo, Wob, DMODEL, DMODEL);

  // 2) fused projections
  gemm_bf16_kernel<<<dim3(QKV_N / 128, MTOK / 128), 256, 0, stream>>>(
      Xb, Wqb, Cqkv, (int)MTOK, QKV_N, DMODEL);
  gemm_bf16_kernel<<<dim3(DMODEL / 128, MTOK / 128), 256, 0, stream>>>(
      Xb, Wgb, Cgat, (int)MTOK, DMODEL, DMODEL);

  // 3) decay scan, 4) xpos rotate + relayout
  gamma_scan_kernel<<<BATCH * H_HEADS, 1024, 0, stream>>>(X, Wg, LB);
  rotate_kernel<<<8192, 256, 0, stream>>>(Cqkv, TS, Qr, Kr, Vr);

  // 5) attention
  attn_kernel<<<dim3(SEQ_N / 128, H_HEADS, BATCH), 256, 0, stream>>>(
      Qr, Kr, Vr, LB, Oat);

  // 6) groupnorm stats + normalize * silu(gated)
  stats_kernel<<<BATCH * H_HEADS, 256, 0, stream>>>(Oat, ST);
  norm_kernel<<<(int)((MTOK * DMODEL + 255) / 256), 256, 0, stream>>>(
      Oat, Cgat, ST, On);

  // 7) output projection
  gemm_bf16_kernel<<<dim3(DMODEL / 128, MTOK / 128), 256, 0, stream>>>(
      On, Wob, out, (int)MTOK, DMODEL, DMODEL);
}
